// CT4LSTM_PPP_74938589381028
// MI455X (gfx1250) — matrix operand order repaired and checked by reading, not compiled, not run on hardware
//
#include <hip/hip_runtime.h>
#include <hip/hip_bf16.h>

typedef _Float16 v16h __attribute__((ext_vector_type(16)));
typedef float    v8f  __attribute__((ext_vector_type(8)));

#define B_   128
#define T_   512
#define S_   511
#define H4_  256
#define NG_  7
#define GN_  (NG_ * H4_)        /* 1792 */
#define NCOMBO_ 12

/* workspace layout */
#define WS_XCOMB   0                    /* 12*64 f32 */
#define WS_IWMP    (WS_XCOMB + 768)     /* 4*64 f32  */
#define WS_COMBO   (WS_IWMP + 256)      /* 12*1792 f32 */
#define WS_F32_TOT (WS_COMBO + 21504)
#define W1T_HALVES (NG_ * 16 * 8 * 512) /* 458752 */
#define W2T_HALVES (NG_ * 4 * 4 * 2 * 512) /* 114688 */

/* ---- branch-free fast transcendentals (v_exp/v_log/v_rcp, no exec divergence) ---- */
__device__ __forceinline__ float sigm(float x) {
  return __builtin_amdgcn_rcpf(1.0f + __expf(-x));
}
__device__ __forceinline__ float tanh_fast(float x) {
  float e = __expf(-2.0f * fabsf(x));                /* e in (0,1] */
  float t = (1.0f - e) * __builtin_amdgcn_rcpf(1.0f + e);
  return copysignf(t, x);
}
__device__ __forceinline__ float softplus_fast(float x) {
  /* max(x,0) + log1p(exp(-|x|)); log argument in [1,2] -> fast log is safe */
  return fmaxf(x, 0.0f) + __logf(1.0f + __expf(-fabsf(x)));
}

/* B-matrix (32x16 f16) half-index -> K mapping: lanes 0-15 hold K=0..7,16..23, lanes 16-31 hold K=8..15,24..31 */
__device__ __forceinline__ int bkB(int lane, int j) { return ((lane < 16) ? 0 : 8) + ((j < 8) ? j : 8 + j); }

/* ---------------- prep: tiny embedding MLPs -> 12 combo vectors + masked intensity weights -------- */
__global__ void prep_tables(const float* __restrict__ etW1, const float* __restrict__ etb1,
                            const float* __restrict__ etW2, const float* __restrict__ etb2,
                            const float* __restrict__ esW1, const float* __restrict__ esb1,
                            const float* __restrict__ esW2, const float* __restrict__ esb2,
                            const float* __restrict__ iW,
                            float* __restrict__ xcomb, float* __restrict__ iWmp) {
  __shared__ float t1e[4][64], t1s[3][64], te[4][64], se[3][64];
  const int tid = threadIdx.x;
  if (tid < 64) {
    for (int e = 0; e < 4; ++e) t1e[e][tid] = tanhf(etW1[tid * 4 + e] + etb1[tid]);
    for (int s = 0; s < 3; ++s) t1s[s][tid] = tanhf(esW1[tid * 3 + s] + esb1[tid]);
  }
  __syncthreads();
  if (tid < 64) {
    for (int e = 0; e < 4; ++e) {
      float a = etb2[tid];
      for (int k = 0; k < 64; ++k) a += t1e[e][k] * etW2[tid * 64 + k];
      te[e][tid] = tanhf(a);
    }
    for (int s = 0; s < 3; ++s) {
      float a = esb2[tid];
      for (int k = 0; k < 64; ++k) a += t1s[s][k] * esW2[tid * 64 + k];
      se[s][tid] = tanhf(a);
    }
  }
  __syncthreads();
  if (tid < 64) {
    for (int c = 0; c < NCOMBO_; ++c) xcomb[c * 64 + tid] = te[c / 3][tid] + se[c % 3][tid];
  }
  for (int idx = tid; idx < 256; idx += 128) {
    int o = idx >> 6, k = idx & 63;
    iWmp[idx] = iW[o * H4_ + o * 64 + k];   /* block-diagonal extract of i_W */
  }
}

/* combo_pre[c][g*256+o] = b1 + sum_i xcomb[c][i] * g_W1[g][o][i] (x-half, i<64) */
__global__ void prep_combo(const float* __restrict__ gW1, const float* __restrict__ gb1,
                           const float* __restrict__ xcomb, float* __restrict__ combo) {
  int idx = blockIdx.x * 256 + threadIdx.x;
  if (idx >= NCOMBO_ * GN_) return;
  int c = idx / GN_, rem = idx % GN_;      /* rem = g*256 + o */
  float s = gb1[rem];
  const float* xv = xcomb + c * 64;
  const float* wr = gW1 + (size_t)rem * 320;
  for (int i = 0; i < 64; ++i) s += xv[i] * wr[i];
  combo[idx] = s;
}

/* W1 h-half -> f16 B-tiles [g][ot(16)][kt(8)][lane(32)][16 halves] */
__global__ void prep_w1(const float* __restrict__ gW1, _Float16* __restrict__ w1t) {
  int idx = blockIdx.x * 256 + threadIdx.x;
  if (idx >= W1T_HALVES) return;
  int half = idx & 511, tile = idx >> 9;
  int lane = half >> 4, j = half & 15;
  int kt = tile & 7, ot = (tile >> 3) & 15, g = tile >> 7;
  int k = kt * 32 + bkB(lane, j);
  int ncol = ot * 16 + (lane & 15);
  w1t[idx] = (_Float16)gW1[((size_t)(g * H4_ + ncol)) * 320 + 64 + k];
}

/* masked W2 diagonal blocks -> f16 B-tiles [g][jb(4)][nb(4)][kt(2)][lane][16] */
__global__ void prep_w2(const float* __restrict__ gW2, _Float16* __restrict__ w2t) {
  int idx = blockIdx.x * 256 + threadIdx.x;
  if (idx >= W2T_HALVES) return;
  int half = idx & 511, tile = idx >> 9;
  int lane = half >> 4, j = half & 15;
  int kt = tile & 1, nb = (tile >> 1) & 3, jb = (tile >> 3) & 3, g = tile >> 5;
  int k = kt * 32 + bkB(lane, j);
  int nloc = nb * 16 + (lane & 15);
  w2t[idx] = (_Float16)gW2[((size_t)(g * H4_ + jb * 64 + nloc)) * H4_ + jb * 64 + k];
}

/* ---------------- main persistent scan: 8 WGs x 16 batch rows, 8 waves each ---------------- */
__global__ __launch_bounds__(256, 1)
void ct4lstm_scan(const int* __restrict__ event, const float* __restrict__ tstep,
                  const int* __restrict__ mstate, const float* __restrict__ combo,
                  const _Float16* __restrict__ w1t, const _Float16* __restrict__ w2t,
                  const float* __restrict__ gb2, const float* __restrict__ iWmp,
                  const float* __restrict__ ib, float* __restrict__ out) {
  extern __shared__ v16h smem_base[];
  _Float16* h16   = (_Float16*)smem_base;   /* 4096 halves : h in WMMA-A tile layout */
  _Float16* pre16 = h16 + 4096;             /* 28672 halves: layer-1 act, A layout    */
  float* h32    = (float*)(pre16 + 28672);  /* 4096 f32 : plain h for intensity head  */
  float* gb2L   = h32 + 4096;               /* 1792 f32 */
  float* comboL = gb2L + GN_;               /* 12*1792 f32 = 86KB, LDS-resident table */
  float* iWmpL  = comboL + NCOMBO_ * GN_;   /* 256 f32 */
  float* dtL    = iWmpL + 256;              /* 16 */
  int*   cidxL  = (int*)(dtL + 16);         /* 16 */

  const int tid  = threadIdx.x;
  const int wave = tid >> 5;
  const int lane = tid & 31;
  const int l15  = lane & 15;
  const int hi8  = (lane >= 16) ? 8 : 0;
  const int rbase = blockIdx.x * 16;

  for (int i = tid; i < 4096; i += 256) { h16[i] = (_Float16)0.0f; h32[i] = 0.0f; }
  for (int i = tid; i < GN_; i += 256) gb2L[i] = gb2[i];
  for (int i = tid; i < NCOMBO_ * GN_; i += 256) comboL[i] = combo[i];
  for (int i = tid; i < 256; i += 256) iWmpL[i] = iWmp[i];
  if (tid < 16) {
    int b = rbase + tid;
    cidxL[tid] = event[b * T_] * 3 + mstate[b * T_];
    dtL[tid]   = tstep[b * T_ + 1];
  }
  __syncthreads();

  /* per-lane constants for the two 16-column groups this wave owns */
  int neq[2], loff[2], jhh[2], kthh[2];
  float gb2v[2][NG_];
#pragma unroll
  for (int q = 0; q < 2; ++q) {
    int ne = (2 * wave + q) * 16 + l15;
    neq[q] = ne;
    int kl = ne & 31, lo, j;                 /* A-layout (m,kl) -> (lane, half) map */
    if (kl < 8)       { lo = 0;  j = kl; }
    else if (kl < 16) { lo = 16; j = kl - 8; }
    else if (kl < 24) { lo = 0;  j = kl - 8; }
    else              { lo = 16; j = kl - 16; }
    loff[q] = lo; jhh[q] = j; kthh[q] = ne >> 5;
    for (int g = 0; g < NG_; ++g) gb2v[q][g] = gb2L[g * H4_ + ne];
  }

  float c_r[2][8] = {{0.f}}, ct_r[2][8] = {{0.f}};

  float* out_int = out;
  float* out_c  = out + (size_t)B_ * S_ * 4;
  float* out_ct = out_c + (size_t)B_ * S_ * H4_;
  float* out_o  = out_ct + (size_t)B_ * S_ * H4_;
  float* out_d  = out_o + (size_t)B_ * S_ * H4_;

  for (int t = 0; t < S_; ++t) {
    int cid[8]; float dtr[8];
#pragma unroll
    for (int r = 0; r < 8; ++r) { int m = r + hi8; cid[r] = cidxL[m]; dtr[r] = dtL[m]; }

    /* ---- layer 1: pre = tanh(h @ W1hT + combo_pre[cid]) ---- */
    for (int g = 0; g < NG_; ++g) {
      v16h a[8];
#pragma unroll
      for (int kt = 0; kt < 8; ++kt)
        a[kt] = *(const v16h*)(h16 + kt * 512 + lane * 16);
#pragma unroll
      for (int q = 0; q < 2; ++q) {
        const int ot = 2 * wave + q;
        v8f acc = {0.f, 0.f, 0.f, 0.f, 0.f, 0.f, 0.f, 0.f};
        const _Float16* wp = w1t + (((size_t)(g * 16 + ot) * 8) << 9) + lane * 16;
#pragma unroll
        for (int kt = 0; kt < 8; ++kt) {
          v16h bm = *(const v16h*)(wp + ((size_t)kt << 9));
          acc = __builtin_amdgcn_wmma_f32_16x16x32_f16(false, a[kt], false, bm,
                                                       (short)0, acc, false, false);
        }
        const int p = g * H4_ + neq[q];
        _Float16* pp = pre16 + (p >> 5) * 512 + jhh[q];
#pragma unroll
        for (int r = 0; r < 8; ++r) {
          const int m = r + hi8;
          float v = acc[r] + comboL[cid[r] * GN_ + p];
          pp[(m + loff[q]) * 16] = (_Float16)tanh_fast(v);
        }
      }
    }
    __syncthreads();

    /* ---- layer 2 (block-diagonal) + continuous-time cell update ---- */
#pragma unroll
    for (int q = 0; q < 2; ++q) {
      const int ot = 2 * wave + q;
      const int jb = ot >> 2, nb = ot & 3;
      const int ne = neq[q];
      v8f ga[NG_];
      for (int g = 0; g < NG_; ++g) {
        v8f acc = {0.f, 0.f, 0.f, 0.f, 0.f, 0.f, 0.f, 0.f};
        const _Float16* wp = w2t + (((size_t)(((g * 4 + jb) * 4 + nb) * 2)) << 9) + lane * 16;
#pragma unroll
        for (int kk = 0; kk < 2; ++kk) {
          const int ktA = g * 8 + jb * 2 + kk;
          v16h av = *(const v16h*)(pre16 + ktA * 512 + lane * 16);
          v16h bm = *(const v16h*)(wp + ((size_t)kk << 9));
          acc = __builtin_amdgcn_wmma_f32_16x16x32_f16(false, av, false, bm,
                                                       (short)0, acc, false, false);
        }
        ga[g] = acc;
      }
      _Float16* hp = h16 + kthh[q] * 512 + jhh[q];
#pragma unroll
      for (int r = 0; r < 8; ++r) {
        const int m = r + hi8;
        float a0 = ga[0][r] + gb2v[q][0];
        float a1 = ga[1][r] + gb2v[q][1];
        float a2 = ga[2][r] + gb2v[q][2];
        float a3 = ga[3][r] + gb2v[q][3];
        float a4 = ga[4][r] + gb2v[q][4];
        float a5 = ga[5][r] + gb2v[q][5];
        float a6 = ga[6][r] + gb2v[q][6];
        float inpt = sigm(a0), forget = sigm(a1), outp = sigm(a2);
        float intg = sigm(a3), fgtg = sigm(a4);
        float z = tanh_fast(a5);
        float decay = softplus_fast(a6);
        float ci  = forget * c_r[q][r] + inpt * z;
        float ctn = fgtg * ct_r[q][r] + intg * z;
        float cn  = ctn + (ci - ctn) * __expf(-decay * dtr[r]);
        float hn  = outp * tanh_fast(cn);
        c_r[q][r] = cn; ct_r[q][r] = ctn;
        const size_t off = ((size_t)(rbase + m) * S_ + t) * H4_ + ne;
        out_c[off] = ci; out_ct[off] = ctn; out_o[off] = outp; out_d[off] = decay;
        h32[m * H4_ + ne] = hn;
        hp[(m + loff[q]) * 16] = (_Float16)hn;
      }
    }

    if (t + 1 < S_ && tid < 16) {           /* prefetch next step's combo idx + dt */
      int b = rbase + tid;
      cidxL[tid] = event[b * T_ + t + 1] * 3 + mstate[b * T_ + t + 1];
      dtL[tid]   = tstep[b * T_ + t + 2];
    }
    __syncthreads();

    /* intensity head: 4 block-diagonal 64-dots, waves 0-1 only (wave-uniform branch) */
    if (tid < 64) {
      const int m = tid & 15, o = tid >> 4;
      float s = ib[o];
      const float* hr = h32 + m * H4_ + o * 64;
      for (int k = 0; k < 64; ++k) s += hr[k] * iWmpL[o * 64 + k];
      out_int[((size_t)(rbase + m) * S_ + t) * 4 + o] = softplus_fast(s);
    }
  }
}

extern "C" void kernel_launch(void* const* d_in, const int* in_sizes, int n_in,
                              void* d_out, int out_size, void* d_ws, size_t ws_size,
                              hipStream_t stream) {
  (void)in_sizes; (void)n_in; (void)out_size; (void)ws_size;
  const int*   event  = (const int*)  d_in[0];
  const float* tstep  = (const float*)d_in[1];
  const int*   mstate = (const int*)  d_in[2];
  const float* etW1 = (const float*)d_in[3];
  const float* etb1 = (const float*)d_in[4];
  const float* etW2 = (const float*)d_in[5];
  const float* etb2 = (const float*)d_in[6];
  const float* esW1 = (const float*)d_in[7];
  const float* esb1 = (const float*)d_in[8];
  const float* esW2 = (const float*)d_in[9];
  const float* esb2 = (const float*)d_in[10];
  const float* gW1  = (const float*)d_in[11];
  const float* gb1  = (const float*)d_in[12];
  const float* gW2  = (const float*)d_in[13];
  const float* gb2  = (const float*)d_in[14];
  const float* iW   = (const float*)d_in[15];
  const float* ib   = (const float*)d_in[16];
  float* out = (float*)d_out;

  float* wsf   = (float*)d_ws;
  float* xcomb = wsf + WS_XCOMB;
  float* iWmp  = wsf + WS_IWMP;
  float* combo = wsf + WS_COMBO;
  _Float16* w1t = (_Float16*)(wsf + WS_F32_TOT);
  _Float16* w2t = w1t + W1T_HALVES;

  prep_tables<<<1, 128, 0, stream>>>(etW1, etb1, etW2, etb2, esW1, esb1, esW2, esb2,
                                     iW, xcomb, iWmp);
  prep_combo<<<(NCOMBO_ * GN_ + 255) / 256, 256, 0, stream>>>(gW1, gb1, xcomb, combo);
  prep_w1<<<(W1T_HALVES + 255) / 256, 256, 0, stream>>>(gW1, w1t);
  prep_w2<<<(W2T_HALVES + 255) / 256, 256, 0, stream>>>(gW2, w2t);

  /* dynamic LDS: h16 + pre16 (f16) + h32 + gb2 + combo table + iWmp + dt + cidx */
  size_t lds = (size_t)4096 * 2 + 28672 * 2
             + ((size_t)4096 + GN_ + NCOMBO_ * GN_ + 256 + 16) * 4 + 16 * 4;
  ct4lstm_scan<<<8, 256, lds, stream>>>(event, tstep, mstate, combo, w1t, w2t,
                                        gb2, iWmp, ib, out);
}
